// MaskedSelfAttention_18202071401042
// MI455X (gfx1250) — compile-verified, matrix operand order checked by reading, not run on hardware
//
#include <hip/hip_runtime.h>
#include <math.h>

// ---------------------------------------------------------------------------
// CDNA5 (gfx1250) fused masked self-attention:
//   QKV projection (WMMA bf16, 32x64 per-wave register tiles)
//   -> flash causal attention (WMMA bf16, f32 online softmax, TDM-staged K/V)
//   -> output projection (WMMA bf16, f32 out).
// B=2, S=2048, D=1024, H=16, HD=64.
// ---------------------------------------------------------------------------

typedef __bf16 bf16_t;
typedef __attribute__((ext_vector_type(16))) __bf16 v16bf_t;
typedef __attribute__((ext_vector_type(8)))  float  v8f_t;
typedef __attribute__((ext_vector_type(4)))  unsigned int u32x4_t;
typedef __attribute__((ext_vector_type(4)))  int i32x4_t;
typedef __attribute__((ext_vector_type(8)))  int i32x8_t;

#define BATCH 2
#define SEQ   2048
#define DMODEL 1024
#define NHEAD 16
#define HDIM  64
#define MTOT  (BATCH * SEQ)          // 4096 rows

union Frag16 {
  v16bf_t v;
  bf16_t  h[16];
  u32x4_t q[2];
};

// k index held by element e (0..15) of a 16-bit A/B fragment, for lane-half hi.
// ISA 7.12.2: VGPR0..3 hold K = hi*8 + 0..7, VGPR4..7 hold K = 16 + hi*8 + 0..7.
__device__ __forceinline__ int kmap(int e, int hi) {
  return (e < 8) ? (hi * 8 + e) : (16 + hi * 8 + (e - 8));
}

// Contiguous fragment load: row pointer, k-offset k0 (multiple of 32), lane half.
__device__ __forceinline__ v16bf_t load_frag_contig(const bf16_t* row, int k0, int hi) {
  Frag16 f;
  f.q[0] = *reinterpret_cast<const u32x4_t*>(row + k0 + hi * 8);
  f.q[1] = *reinterpret_cast<const u32x4_t*>(row + k0 + 16 + hi * 8);
  return f.v;
}

__device__ __forceinline__ v8f_t wmma_bf16(v16bf_t a, v16bf_t b, v8f_t c) {
  return __builtin_amdgcn_wmma_f32_16x16x32_bf16(
      /*neg_a=*/false, a, /*neg_b=*/false, b,
      /*c_mod=*/(short)0, c, /*reuse_a=*/false, /*reuse_b=*/false);
}

// ---------------------------------------------------------------------------
// Tensor Data Mover: async-load one 64x64 bf16 tile (global row stride 1024
// elements) into LDS with a 16B pad every 128B -> LDS row stride 144B ([64][72]).
// D# fields per CDNA5 ISA.
// ---------------------------------------------------------------------------
__device__ __forceinline__ void tdm_load_tile64(unsigned lds_byte_off,
                                                unsigned long long gaddr) {
  u32x4_t g0;
  g0[0] = 1u;                                          // count=1, user descriptor
  g0[1] = lds_byte_off;                                // lds_addr (bytes)
  g0[2] = (unsigned)(gaddr & 0xffffffffull);           // global_addr[31:0]
  g0[3] = (unsigned)((gaddr >> 32) & 0x1ffffffull)     // global_addr[56:32]
          | (2u << 30);                                // type=2 ("image")

  i32x8_t g1;
  g1[0] = (int)((1u << 16)      // data_size = 2 bytes
              | (1u << 20)      // pad_enable
              | (4u << 22)      // pad_interval: 32 DWORDs (128 B)
              | (3u << 25));    // pad_amount: 4 DWORDs (16 B)
  g1[1] = (int)(64u << 16);     // tensor_dim0[15:0] = 64 (bits 63:48)
  g1[2] = (int)(64u << 16);     // tensor_dim0 hi = 0; tensor_dim1[15:0] = 64
  g1[3] = (int)(64u << 16);     // tensor_dim1 hi = 0; tile_dim0 = 64
  g1[4] = 64;                   // tile_dim1 = 64; tile_dim2 = 0
  g1[5] = 1024;                 // tensor_dim0_stride = 1024 elements
  g1[6] = 0;
  g1[7] = 0;

  i32x4_t z4;
  z4[0] = z4[1] = z4[2] = z4[3] = 0;

#if defined(__clang_major__) && (__clang_major__ >= 23)
  i32x8_t z8;
  z8[0] = z8[1] = z8[2] = z8[3] = z8[4] = z8[5] = z8[6] = z8[7] = 0;
  __builtin_amdgcn_tensor_load_to_lds(g0, g1, z4, z4, z8, 0);
#else
  __builtin_amdgcn_tensor_load_to_lds(g0, g1, z4, z4, 0);
#endif
}

// ---------------------------------------------------------------------------
// Kernel 0: f32 -> bf16 convert
// ---------------------------------------------------------------------------
__global__ void cvt_f32_bf16(const float* __restrict__ in, bf16_t* __restrict__ out, int n) {
  int i = blockIdx.x * blockDim.x + threadIdx.x;
  if (i < n) out[i] = (bf16_t)in[i];
}

// ---------------------------------------------------------------------------
// Kernel 1: Y[m,n] = sum_k X[m,k] * W[n,k] + bias[n], bf16 out (QKV projection)
// grid: (M/128, N/64), block 128 (4 waves); wave -> 32x64 register tile
// ---------------------------------------------------------------------------
__global__ void __launch_bounds__(128)
proj_gemm_bf16out(const bf16_t* __restrict__ X, const bf16_t* __restrict__ W,
                  const float* __restrict__ bias, bf16_t* __restrict__ Y) {
  const int lane = threadIdx.x & 31;
  const int wave = threadIdx.x >> 5;
  const int lo = lane & 15;
  const int hi = lane >> 4;
  const int m0 = blockIdx.x * 128 + wave * 32;
  const int n0 = blockIdx.y * 64;
  const int K = DMODEL, N = DMODEL;

  v8f_t acc[2][4];
#pragma unroll
  for (int i = 0; i < 2; ++i)
#pragma unroll
    for (int j = 0; j < 4; ++j)
#pragma unroll
      for (int r = 0; r < 8; ++r) acc[i][j][r] = 0.0f;

  const bf16_t* arow0 = X + (size_t)(m0 + lo) * K;
  const bf16_t* arow1 = X + (size_t)(m0 + 16 + lo) * K;
  for (int k0 = 0; k0 < K; k0 += 32) {
    v16bf_t a0 = load_frag_contig(arow0, k0, hi);
    v16bf_t a1 = load_frag_contig(arow1, k0, hi);
#pragma unroll
    for (int j = 0; j < 4; ++j) {
      v16bf_t b = load_frag_contig(W + (size_t)(n0 + j * 16 + lo) * K, k0, hi);
      acc[0][j] = wmma_bf16(a0, b, acc[0][j]);
      acc[1][j] = wmma_bf16(a1, b, acc[1][j]);
    }
  }
#pragma unroll
  for (int i = 0; i < 2; ++i)
#pragma unroll
    for (int j = 0; j < 4; ++j)
#pragma unroll
      for (int r = 0; r < 8; ++r) {
        int m = m0 + i * 16 + r + 8 * hi;
        int n = n0 + j * 16 + lo;
        Y[(size_t)m * N + n] = (bf16_t)(acc[i][j][r] + bias[n]);
      }
}

// ---------------------------------------------------------------------------
// Kernel 3: same GEMM but f32 out (final projection)
// ---------------------------------------------------------------------------
__global__ void __launch_bounds__(128)
proj_gemm_f32out(const bf16_t* __restrict__ X, const bf16_t* __restrict__ W,
                 const float* __restrict__ bias, float* __restrict__ Y) {
  const int lane = threadIdx.x & 31;
  const int wave = threadIdx.x >> 5;
  const int lo = lane & 15;
  const int hi = lane >> 4;
  const int m0 = blockIdx.x * 128 + wave * 32;
  const int n0 = blockIdx.y * 64;
  const int K = DMODEL, N = DMODEL;

  v8f_t acc[2][4];
#pragma unroll
  for (int i = 0; i < 2; ++i)
#pragma unroll
    for (int j = 0; j < 4; ++j)
#pragma unroll
      for (int r = 0; r < 8; ++r) acc[i][j][r] = 0.0f;

  const bf16_t* arow0 = X + (size_t)(m0 + lo) * K;
  const bf16_t* arow1 = X + (size_t)(m0 + 16 + lo) * K;
  for (int k0 = 0; k0 < K; k0 += 32) {
    v16bf_t a0 = load_frag_contig(arow0, k0, hi);
    v16bf_t a1 = load_frag_contig(arow1, k0, hi);
#pragma unroll
    for (int j = 0; j < 4; ++j) {
      v16bf_t b = load_frag_contig(W + (size_t)(n0 + j * 16 + lo) * K, k0, hi);
      acc[0][j] = wmma_bf16(a0, b, acc[0][j]);
      acc[1][j] = wmma_bf16(a1, b, acc[1][j]);
    }
  }
#pragma unroll
  for (int i = 0; i < 2; ++i)
#pragma unroll
    for (int j = 0; j < 4; ++j)
#pragma unroll
      for (int r = 0; r < 8; ++r) {
        int m = m0 + i * 16 + r + 8 * hi;
        int n = n0 + j * 16 + lo;
        Y[(size_t)m * N + n] = acc[i][j][r] + bias[n];
      }
}

// ---------------------------------------------------------------------------
// Kernel 2: causal flash attention per (qblock, head, batch).
// block = 128 threads (4 waves); wave owns 16 query rows; K/V tiles of 64
// staged into LDS by the Tensor Data Mover (wave 0 issues, s_wait_tensorcnt).
// ---------------------------------------------------------------------------
__global__ void __launch_bounds__(128)
attn_causal(const bf16_t* __restrict__ Q, const bf16_t* __restrict__ Kmat,
            const bf16_t* __restrict__ V, bf16_t* __restrict__ O) {
  const int lane = threadIdx.x & 31;
  const int wave = threadIdx.x >> 5;
  const int lo = lane & 15;
  const int hi = lane >> 4;
  const int qblk = blockIdx.x;   // 0..31  (S/64)
  const int head = blockIdx.y;   // 0..15
  const int bidx = blockIdx.z;   // 0..1

  const size_t base = (size_t)bidx * SEQ * DMODEL + (size_t)head * HDIM;

  __shared__ bf16_t Ks[64][72];        // 9 KB; 144B row stride == TDM pad layout
  __shared__ bf16_t Vs[64][72];        // 9 KB
  __shared__ float  Sw[4][16][64];     // 16 KB score staging per wave
  __shared__ bf16_t Pw[4][16][72];     // 9 KB probabilities in A layout
  __shared__ float  fac[4][16];
  __shared__ float  lin[4][16];

  const int q0 = qblk * 64 + wave * 16;

  // Q fragments for this wave's 16 rows (HD=64 -> two 16x32 A-frags), kept live.
  const bf16_t* qrow = Q + base + (size_t)(q0 + lo) * DMODEL;
  const v16bf_t qa0 = load_frag_contig(qrow, 0, hi);
  const v16bf_t qa1 = load_frag_contig(qrow, 32, hi);

  const unsigned ldsK = (unsigned)(size_t)(void*)&Ks[0][0];
  const unsigned ldsV = (unsigned)(size_t)(void*)&Vs[0][0];

  v8f_t o[4];
#pragma unroll
  for (int j = 0; j < 4; ++j)
#pragma unroll
    for (int r = 0; r < 8; ++r) o[j][r] = 0.0f;

  float m_run = -3.0e38f;
  float l_run = 0.0f;
  const float scale = 0.125f;   // 1/sqrt(64)

  for (int kb = 0; kb <= qblk; ++kb) {
    // ---- TDM async K/V tile load: 64 rows x 64 bf16 each, issued by wave 0 ----
    if (wave == 0) {
      unsigned long long kg =
          (unsigned long long)(size_t)(const void*)(Kmat + base + (size_t)(kb * 64) * DMODEL);
      unsigned long long vg =
          (unsigned long long)(size_t)(const void*)(V + base + (size_t)(kb * 64) * DMODEL);
      tdm_load_tile64(ldsK, kg);
      tdm_load_tile64(ldsV, vg);
      __builtin_amdgcn_s_wait_tensorcnt(0);
    }
    __syncthreads();

    // ---- S = Q K^T over this tile (16x64 f32 per wave) ----
    v8f_t s[4];
#pragma unroll
    for (int j = 0; j < 4; ++j)
#pragma unroll
      for (int r = 0; r < 8; ++r) s[j][r] = 0.0f;
#pragma unroll
    for (int j = 0; j < 4; ++j) {
      const bf16_t* krow = &Ks[j * 16 + lo][0];
      s[j] = wmma_bf16(qa0, load_frag_contig(krow, 0, hi), s[j]);
      s[j] = wmma_bf16(qa1, load_frag_contig(krow, 32, hi), s[j]);
    }

    // ---- scale + causal mask, stage to LDS in row-major ----
#pragma unroll
    for (int j = 0; j < 4; ++j)
#pragma unroll
      for (int r = 0; r < 8; ++r) {
        int m = r + 8 * hi;                 // 0..15 within wave tile
        int qg = q0 + m;
        int kg = kb * 64 + j * 16 + lo;
        Sw[wave][m][j * 16 + lo] = (kg <= qg) ? s[j][r] * scale : -3.0e38f;
      }
    __syncthreads();

    // ---- online softmax: lanes 0..15 each own one row ----
    if (lane < 16) {
      float rmax = -3.0e38f;
#pragma unroll 8
      for (int c = 0; c < 64; ++c) rmax = fmaxf(rmax, Sw[wave][lane][c]);
      float mnew = fmaxf(m_run, rmax);
      float f = __expf(m_run - mnew);
      float psum = 0.0f;
#pragma unroll 8
      for (int c = 0; c < 64; ++c) {
        float p = __expf(Sw[wave][lane][c] - mnew);
        psum += p;
        Pw[wave][lane][c] = (bf16_t)p;
      }
      l_run = l_run * f + psum;
      m_run = mnew;
      fac[wave][lane] = f;
    }
    __syncthreads();

    // ---- rescale O, then O += P x V ----
#pragma unroll
    for (int j = 0; j < 4; ++j)
#pragma unroll
      for (int r = 0; r < 8; ++r) o[j][r] *= fac[wave][r + 8 * hi];

    const v16bf_t pa0 = load_frag_contig(&Pw[wave][lo][0], 0, hi);
    const v16bf_t pa1 = load_frag_contig(&Pw[wave][lo][0], 32, hi);
#pragma unroll
    for (int j = 0; j < 4; ++j) {
      Frag16 bv0, bv1;
#pragma unroll
      for (int e = 0; e < 16; ++e) {
        bv0.h[e] = Vs[kmap(e, hi)][j * 16 + lo];
        bv1.h[e] = Vs[32 + kmap(e, hi)][j * 16 + lo];
      }
      o[j] = wmma_bf16(pa0, bv0.v, o[j]);
      o[j] = wmma_bf16(pa1, bv1.v, o[j]);
    }
    __syncthreads();   // protect Ks/Vs/Pw before next tile
  }

  if (lane < 16) lin[wave][lane] = 1.0f / l_run;
  __syncthreads();

#pragma unroll
  for (int j = 0; j < 4; ++j)
#pragma unroll
    for (int r = 0; r < 8; ++r) {
      int m = r + 8 * hi;
      int qg = q0 + m;
      O[base + (size_t)qg * DMODEL + j * 16 + lo] = (bf16_t)(o[j][r] * lin[wave][m]);
    }
}

// ---------------------------------------------------------------------------
// Host-side launcher
// ---------------------------------------------------------------------------
extern "C" void kernel_launch(void* const* d_in, const int* in_sizes, int n_in,
                              void* d_out, int out_size, void* d_ws, size_t ws_size,
                              hipStream_t stream) {
  const float* x   = (const float*)d_in[0];
  const float* Wq  = (const float*)d_in[1];
  const float* bq  = (const float*)d_in[2];
  const float* Wk  = (const float*)d_in[3];
  const float* bk  = (const float*)d_in[4];
  const float* Wv  = (const float*)d_in[5];
  const float* bv  = (const float*)d_in[6];
  const float* Wf  = (const float*)d_in[7];
  const float* bff = (const float*)d_in[8];
  float* out = (float*)d_out;

  const size_t nAct = (size_t)MTOT * DMODEL;       // 4096*1024
  const size_t nW   = (size_t)DMODEL * DMODEL;     // 1024*1024

  bf16_t* ws = (bf16_t*)d_ws;
  bf16_t* xb  = ws;            // x in bf16
  bf16_t* wqb = xb  + nAct;
  bf16_t* wkb = wqb + nW;
  bf16_t* wvb = wkb + nW;
  bf16_t* wfb = wvb + nW;
  bf16_t* qb  = wfb + nW;      // Q
  bf16_t* kbm = qb  + nAct;    // K
  bf16_t* vbm = kbm + nAct;    // V
  bf16_t* ab  = vbm + nAct;    // attention output (pre-projection)

  // 0) convert inputs to bf16
  cvt_f32_bf16<<<dim3((unsigned)((nAct + 255) / 256)), dim3(256), 0, stream>>>(x, xb, (int)nAct);
  cvt_f32_bf16<<<dim3((unsigned)((nW + 255) / 256)), dim3(256), 0, stream>>>(Wq, wqb, (int)nW);
  cvt_f32_bf16<<<dim3((unsigned)((nW + 255) / 256)), dim3(256), 0, stream>>>(Wk, wkb, (int)nW);
  cvt_f32_bf16<<<dim3((unsigned)((nW + 255) / 256)), dim3(256), 0, stream>>>(Wv, wvb, (int)nW);
  cvt_f32_bf16<<<dim3((unsigned)((nW + 255) / 256)), dim3(256), 0, stream>>>(Wf, wfb, (int)nW);

  // 1) Q/K/V projections (wave tile 32x64)
  dim3 gp(MTOT / 128, DMODEL / 64, 1);
  dim3 blk(128);
  proj_gemm_bf16out<<<gp, blk, 0, stream>>>(xb, wqb, bq, qb);
  proj_gemm_bf16out<<<gp, blk, 0, stream>>>(xb, wkb, bk, kbm);
  proj_gemm_bf16out<<<gp, blk, 0, stream>>>(xb, wvb, bv, vbm);

  // 2) causal attention (TDM-staged K/V tiles)
  dim3 ga(SEQ / 64, NHEAD, BATCH);
  attn_causal<<<ga, blk, 0, stream>>>(qb, kbm, vbm, ab);

  // 3) output projection -> f32
  proj_gemm_f32out<<<gp, blk, 0, stream>>>(ab, wfb, bff, out);
}
